// FastRCNNConvFCHead_RM_51608327029336
// MI455X (gfx1250) — matrix operand order repaired and checked by reading, not compiled, not run on hardware
//
#include <hip/hip_runtime.h>
#include <stdint.h>

typedef __attribute__((ext_vector_type(16))) __bf16 v16bf;
typedef __attribute__((ext_vector_type(8)))  __bf16 v8bf;
typedef __attribute__((ext_vector_type(8)))  float  v8f;

#define SKE 72                    // padded LDS row stride (elements): 144B, 16B-aligned, bank-conflict-free
#define TILE_ELEMS (128 * SKE)    // one 128x64 bf16 tile (padded)

__device__ __forceinline__ v16bf cat8(v8bf lo, v8bf hi) {
  return __builtin_shufflevector(lo, hi, 0,1,2,3,4,5,6,7,8,9,10,11,12,13,14,15);
}

// C[M,N] = A[M,K] (row-major bf16) x B[N,K]^T (row-major bf16) + bias, batched.
// BK=64, register-staged double-buffered LDS (one barrier per K-tile),
// compute via v_wmma_f32_16x16x32_bf16 with f32 accumulators.
__global__ __launch_bounds__(256)
void gemm_bf16_wmma(const __bf16* __restrict__ A, const __bf16* __restrict__ B,
                    const float* __restrict__ bias,
                    float* __restrict__ outF, __bf16* __restrict__ outB,
                    int M, int N, int K, int lda, int ldb, int ldc,
                    long strA, long strB, long strBias, long strC)
{
  // [buf][mat]: buf0A, buf0B, buf1A, buf1B  (total 72 KB)
  __shared__ __attribute__((aligned(16))) __bf16 sh[4][TILE_ELEMS];

  const int tid  = threadIdx.x;
  const int lane = tid & 31;
  const int wid  = tid >> 5;
  const int wm = (wid & 3) * 32;   // 4 waves along M
  const int wn = (wid >> 2) * 64;  // 2 waves along N
  const int bm = blockIdx.y * 128;
  const int bn = blockIdx.x * 128;
  const int z  = blockIdx.z;

  const __bf16* Ap = A + (long)z * strA + (long)bm * lda;
  const __bf16* Bp = B + (long)z * strB + (long)bn * ldb;

  const v8f vzero = {0.f,0.f,0.f,0.f,0.f,0.f,0.f,0.f};
  v8f acc[2][4];
#pragma unroll
  for (int i = 0; i < 2; ++i)
#pragma unroll
    for (int j = 0; j < 4; ++j) acc[i][j] = vzero;

  // copy mapping: 128 rows x 64 cols, 16B chunks; 4 chunks/matrix/thread
  const int rr = tid >> 3;          // 0..31 (+32*it)
  const int kc = (tid & 7) * 8;     // 0..56
  v8bf z8;
#pragma unroll
  for (int e = 0; e < 8; ++e) z8[e] = (__bf16)0.0f;

  v8bf ra[4], rb[4];                // register staging for the next tile

  auto gload = [&](int k0) {
#pragma unroll
    for (int it = 0; it < 4; ++it) {
      const int r = rr + it*32;
      ra[it] = *(const v8bf*)&Ap[(long)r * lda + k0 + kc];
      rb[it] = z8;
      if (bn + r < N) rb[it] = *(const v8bf*)&Bp[(long)r * ldb + k0 + kc];
    }
    if (k0 + 64 < K) {   // warm tile t+2 (global_prefetch_b8)
      __builtin_prefetch(&Ap[(long)rr * lda + k0 + 64 + kc], 0, 1);
      if (bn + rr < N) __builtin_prefetch(&Bp[(long)rr * ldb + k0 + 64 + kc], 0, 1);
    }
  };
  auto lstore = [&](int b) {
    __bf16* dA = sh[2*b + 0];
    __bf16* dB = sh[2*b + 1];
#pragma unroll
    for (int it = 0; it < 4; ++it) {
      const int r = rr + it*32;
      *(v8bf*)&dA[r * SKE + kc] = ra[it];
      *(v8bf*)&dB[r * SKE + kc] = rb[it];
    }
  };

  const int T = K >> 6;   // all K used here are multiples of 64
  gload(0);
  lstore(0);

  for (int t = 0; t < T; ++t) {
    const int cur = t & 1;
    if (t + 1 < T) gload((t + 1) << 6);   // in flight during barrier + compute
    __syncthreads();                       // publishes lstore(t); protects buf reuse

    const __bf16* tA = sh[2*cur + 0];
    const __bf16* tB = sh[2*cur + 1];
    const int hh = lane >> 4;
    const int fr = lane & 15;
#pragma unroll
    for (int ks = 0; ks < 2; ++ks) {
      v16bf af[2], bfv[4];
#pragma unroll
      for (int i = 0; i < 2; ++i) {
        // A 16-bit 16x32 layout: lanes 0-15 hold K 0..7 & 16..23; lanes 16-31 hold K 8..15 & 24..31
        const __bf16* p = &tA[(wm + i*16 + fr) * SKE + ks*32];
        af[i] = cat8(*(const v8bf*)(p + hh*8), *(const v8bf*)(p + 16 + hh*8));
      }
#pragma unroll
      for (int j = 0; j < 4; ++j) {
        // B 16-bit 32x16 layout: lanes 0-15 hold K 0..7 & 16..23; lanes 16-31 hold K 8..15 & 24..31
        const __bf16* p = &tB[(wn + j*16 + fr) * SKE + ks*32];
        bfv[j] = cat8(*(const v8bf*)(p + hh*8), *(const v8bf*)(p + 16 + hh*8));
      }
#pragma unroll
      for (int i = 0; i < 2; ++i)
#pragma unroll
        for (int j = 0; j < 4; ++j)
          acc[i][j] = __builtin_amdgcn_wmma_f32_16x16x32_bf16(
              false, af[i], false, bfv[j], (short)0, acc[i][j], false, false);
    }
    if (t + 1 < T) lstore(cur ^ 1);  // safe: all waves passed this iter's barrier
  }

  // ---- epilogue. C layout: lane = (mHalf<<4)|n ; VGPR e -> row mHalf*8+e
  const int hh = lane >> 4;
  const int fr = lane & 15;
#pragma unroll
  for (int j = 0; j < 4; ++j) {
    const int n = bn + wn + j*16 + fr;
    if (n >= N) continue;
    const float bv = bias ? bias[(long)z * strBias + n] : 0.0f;
#pragma unroll
    for (int i = 0; i < 2; ++i) {
      const int mb = bm + wm + i*16 + hh*8;
#pragma unroll
      for (int e = 0; e < 8; ++e) {
        const float v = acc[i][j][e] + bv;
        const long idx = (long)z * strC + (long)(mb + e) * ldc + n;
        if (outF) outF[idx] = v;
        if (outB) outB[idx] = (__bf16)v;
      }
    }
  }
}

// out[C][R] (bf16) = transpose of in[R][C] (f32)
__global__ __launch_bounds__(256)
void transpose_f32_to_bf16(const float* __restrict__ in, __bf16* __restrict__ out, int R, int C)
{
  __shared__ float t[32][33];
  const int c = blockIdx.x*32 + threadIdx.x;
#pragma unroll
  for (int j = 0; j < 4; ++j) {
    const int r = blockIdx.y*32 + threadIdx.y + j*8;
    if (r < R && c < C) t[threadIdx.y + j*8][threadIdx.x] = in[(long)r*C + c];
  }
  __syncthreads();
  const int r2 = blockIdx.y*32 + threadIdx.x;
#pragma unroll
  for (int j = 0; j < 4; ++j) {
    const int c2 = blockIdx.x*32 + threadIdx.y + j*8;
    if (c2 < C && r2 < R) out[(long)c2*R + r2] = (__bf16)t[threadIdx.x][threadIdx.y + j*8];
  }
}

__global__ __launch_bounds__(256)
void cvt_f32_to_bf16(const float* __restrict__ in, __bf16* __restrict__ out, long n)
{
  const long i = (long)blockIdx.x * 256 + threadIdx.x;
  if (i < n) out[i] = (__bf16)in[i];
}

// W[g][n][m] = 0.125*W[g][n][m] + log(max(relu(pe(n,m) . wg_w[:,g] + wg_b[g]), 1e-6))
// pe computed on the fly from boxes (fuses _pos_matrix + _pos_embedding + wg projection).
__global__ __launch_bounds__(256)
void posbias_kernel(const float* __restrict__ boxes, const float* __restrict__ wgw,
                    const float* __restrict__ wgb, float* __restrict__ W)
{
  __shared__ float swg[64*16];
  __shared__ float swb[16];
  const int tid = threadIdx.x;
  for (int i = tid; i < 1024; i += 256) swg[i] = wgw[i];
  if (tid < 16) swb[tid] = wgb[tid];
  __syncthreads();

  const int n = blockIdx.y;
  const int m = blockIdx.x*256 + tid;

  const float nx0 = boxes[n*4+0], ny0 = boxes[n*4+1], nx1 = boxes[n*4+2], ny1 = boxes[n*4+3];
  const float wn_ = nx1-nx0, hn_ = ny1-ny0;
  const float cxn = nx0 + 0.5f*wn_, cyn = ny0 + 0.5f*hn_;
  const float mx0 = boxes[m*4+0], my0 = boxes[m*4+1], mx1 = boxes[m*4+2], my1 = boxes[m*4+3];
  const float wm_ = mx1-mx0, hm_ = my1-my0;
  const float cxm = mx0 + 0.5f*wm_, cym = my0 + 0.5f*hm_;

  float p4[4];
  p4[0] = logf(fmaxf(fabsf(cxm-cxn)/wm_, 0.001f));
  p4[1] = logf(fmaxf(fabsf(cym-cyn)/hm_, 0.001f));
  p4[2] = logf(wm_/wn_);
  p4[3] = logf(hm_/hn_);

  float acc[16];
#pragma unroll
  for (int g = 0; g < 16; ++g) acc[g] = swb[g];

  // 100 / 1000^(f/8)
  const float SC[8] = {100.0f, 42.169650f, 17.782794f, 7.4989421f,
                       3.1622777f, 1.3335214f, 0.56234133f, 0.23713737f};
#pragma unroll
  for (int f = 0; f < 8; ++f) {
#pragma unroll
    for (int c = 0; c < 4; ++c) {
      const float d = p4[c] * SC[f];
      const float s = __sinf(d), co = __cosf(d);
      const float* ws = &swg[(c*16 + f)     * 16];
      const float* wc = &swg[(c*16 + 8 + f) * 16];
#pragma unroll
      for (int g = 0; g < 16; ++g)
        acc[g] = fmaf(s, ws[g], fmaf(co, wc[g], acc[g]));
    }
  }
  const long idx = (long)n*1024 + m;
#pragma unroll
  for (int g = 0; g < 16; ++g) {
    float v = fmaxf(acc[g], 0.0f);
    v = logf(fmaxf(v, 1e-6f));
    const long o = (long)g*1048576 + idx;
    W[o] = W[o]*0.125f + v;   // fold 1/sqrt(DG) scale of aff here
  }
}

// softmax over last dim (1024) of W[16][1024][1024]; output bf16 p
__global__ __launch_bounds__(256)
void softmax_kernel(const float* __restrict__ W, __bf16* __restrict__ P)
{
  __shared__ float red[256];
  const long basei = (long)blockIdx.x * 1024;
  const int tid = threadIdx.x;
  float v[4];
  float mx = -3.402823466e38f;
#pragma unroll
  for (int e = 0; e < 4; ++e) { v[e] = W[basei + tid + e*256]; mx = fmaxf(mx, v[e]); }
  red[tid] = mx; __syncthreads();
  for (int s = 128; s > 0; s >>= 1) { if (tid < s) red[tid] = fmaxf(red[tid], red[tid+s]); __syncthreads(); }
  const float m = red[0]; __syncthreads();
  float sum = 0.0f;
#pragma unroll
  for (int e = 0; e < 4; ++e) { v[e] = __expf(v[e] - m); sum += v[e]; }
  red[tid] = sum; __syncthreads();
  for (int s = 128; s > 0; s >>= 1) { if (tid < s) red[tid] += red[tid+s]; __syncthreads(); }
  const float inv = 1.0f / red[0];
#pragma unroll
  for (int e = 0; e < 4; ++e) P[basei + tid + e*256] = (__bf16)(v[e] * inv);
}

// x = relu(xin + att); write f32 (and optionally bf16 mirror)
__global__ __launch_bounds__(256)
void combine_relu(const float* __restrict__ xin, const float* __restrict__ att,
                  float* __restrict__ outF, __bf16* __restrict__ outB, int n)
{
  const int i = blockIdx.x*256 + threadIdx.x;
  if (i < n) {
    const float v = fmaxf(xin[i] + att[i], 0.0f);
    outF[i] = v;
    if (outB) outB[i] = (__bf16)v;
  }
}

static inline void launch_gemm(hipStream_t s, const __bf16* A, const __bf16* B,
                               const float* bias, float* outF, __bf16* outB,
                               int M, int N, int K, int lda, int ldb, int ldc,
                               long sa, long sb, long sbias, long sc, int batch)
{
  dim3 grid((N + 127)/128, M/128, batch);
  gemm_bf16_wmma<<<grid, 256, 0, s>>>(A, B, bias, outF, outB,
                                      M, N, K, lda, ldb, ldc, sa, sb, sbias, sc);
}

extern "C" void kernel_launch(void* const* d_in, const int* in_sizes, int n_in,
                              void* d_out, int out_size, void* d_ws, size_t ws_size,
                              hipStream_t stream)
{
  (void)in_sizes; (void)n_in; (void)out_size; (void)ws_size;
  const float* box_features = (const float*)d_in[0];
  const float* pred_boxes   = (const float*)d_in[1];
  const float* fc1_w = (const float*)d_in[2];
  const float* fc1_b = (const float*)d_in[3];
  const float* fc2_w = (const float*)d_in[4];
  const float* fc2_b = (const float*)d_in[5];
  const float* wg_w  = (const float*)d_in[6];
  const float* wg_b  = (const float*)d_in[7];
  const float* wq_w  = (const float*)d_in[8];
  const float* wq_b  = (const float*)d_in[9];
  const float* wk_w  = (const float*)d_in[10];
  const float* wk_b  = (const float*)d_in[11];
  const float* wv_w  = (const float*)d_in[12];
  const float* wv_b  = (const float*)d_in[13];
  float* out = (float*)d_out;

  // ---- workspace carving (256B aligned)
  char* base = (char*)d_ws;
  size_t off = 0;
  auto alloc = [&](size_t bytes) -> char* {
    char* p = base + off;
    off += (bytes + 255) & ~(size_t)255;
    return p;
  };
  __bf16* bfA  = (__bf16*)alloc((size_t)2048*12544*2); // box_features bf16
  __bf16* w1T  = (__bf16*)alloc((size_t)12544*1024*2); // fc1_w^T
  __bf16* w2T  = (__bf16*)alloc((size_t)1024*1024*2);  // fc2_w^T
  __bf16* wqT  = (__bf16*)alloc((size_t)2*1024*1024*2);
  __bf16* wkT  = (__bf16*)alloc((size_t)2*1024*1024*2);
  __bf16* wvB  = (__bf16*)alloc((size_t)2*1024*1024*2); // wv_w bf16 (natural [g][o][d])
  float*  xF0  = (float*) alloc((size_t)2048*1024*4);
  __bf16* xB0  = (__bf16*)alloc((size_t)2048*1024*2);
  float*  xF1  = (float*) alloc((size_t)2048*1024*4);
  __bf16* xB1  = (__bf16*)alloc((size_t)2048*1024*2);
  __bf16* qB   = (__bf16*)alloc((size_t)1024*1024*2);
  __bf16* kB   = (__bf16*)alloc((size_t)1024*1024*2);
  __bf16* vT   = (__bf16*)alloc((size_t)1024*1024*2);
  float*  Wf   = (float*) alloc((size_t)16*1024*1024*4); // logits [g][n][m]
  __bf16* pB   = (__bf16*)alloc((size_t)16*1024*1024*2); // softmax probs
  __bf16* o1B  = (__bf16*)alloc((size_t)16*1024*1024*2); // p @ V
  float*  attF = (float*) alloc((size_t)1024*1024*4);

  const dim3 tb32(32, 8, 1);

  // ---- weight / input prep (bf16 + transposes so every GEMM B is [N][K])
  {
    long n = (long)2048*12544;
    cvt_f32_to_bf16<<<(unsigned)((n + 255)/256), 256, 0, stream>>>(box_features, bfA, n);
  }
  transpose_f32_to_bf16<<<dim3(32, 392), tb32, 0, stream>>>(fc1_w, w1T, 12544, 1024);
  transpose_f32_to_bf16<<<dim3(32, 32),  tb32, 0, stream>>>(fc2_w, w2T, 1024, 1024);
  for (int l = 0; l < 2; ++l) {
    transpose_f32_to_bf16<<<dim3(32, 32), tb32, 0, stream>>>(wq_w + (size_t)l*1048576, wqT + (size_t)l*1048576, 1024, 1024);
    transpose_f32_to_bf16<<<dim3(32, 32), tb32, 0, stream>>>(wk_w + (size_t)l*1048576, wkT + (size_t)l*1048576, 1024, 1024);
  }
  {
    long n = (long)2*1048576;
    cvt_f32_to_bf16<<<(unsigned)((n + 255)/256), 256, 0, stream>>>(wv_w, wvB, n);
  }

  // ---- fc1: x = box_features @ fc1_w + b   [2048 x 1024]
  launch_gemm(stream, bfA, w1T, fc1_b, xF0, xB0,
              2048, 1024, 12544, 12544, 12544, 1024, 0, 0, 0, 0, 1);

  for (int l = 0; l < 2; ++l) {
    float*  xF = l ? xF1 : xF0;
    __bf16* xB = l ? xB1 : xB0;
    for (int i = 0; i < 2; ++i) {
      const __bf16* xiB = xB + (size_t)i*1048576;
      float*        xiF = xF + (size_t)i*1048576;

      // q / k projections -> bf16 [n][g*64+d] / [m][g*64+d]
      launch_gemm(stream, xiB, wqT + (size_t)l*1048576, wq_b + (size_t)l*1024,
                  nullptr, qB, 1024, 1024, 1024, 1024, 1024, 1024, 0, 0, 0, 0, 1);
      launch_gemm(stream, xiB, wkT + (size_t)l*1048576, wk_b + (size_t)l*1024,
                  nullptr, kB, 1024, 1024, 1024, 1024, 1024, 1024, 0, 0, 0, 0, 1);

      // aff[g][n][m] = q_g . k_g^T   (batched over 16 groups, K=64)
      launch_gemm(stream, qB, kB, nullptr, Wf, nullptr,
                  1024, 1024, 64, 1024, 1024, 1024, 64, 64, 0, 1048576, 16);

      // W = W/8 + log(max(relu(pe @ wg + b),1e-6)) fused, then softmax -> p bf16
      posbias_kernel<<<dim3(4, 1024), 256, 0, stream>>>(
          pred_boxes + (size_t)i*4096, wg_w + (size_t)l*1024, wg_b + (size_t)l*16, Wf);
      softmax_kernel<<<16384, 256, 0, stream>>>(Wf, pB);

      // V^T (bf16 [d][m]) from current x
      transpose_f32_to_bf16<<<dim3(32, 32), tb32, 0, stream>>>(xiF, vT, 1024, 1024);

      // out1[g][n][d] = p_g @ V   (batched, B shared)
      launch_gemm(stream, pB, vT, nullptr, nullptr, o1B,
                  1024, 1024, 1024, 1024, 1024, 1024, 1048576, 0, 0, 1048576, 16);

      // att[n][g*64+o] = out1_g @ wv_g^T + wv_b  (N=64 per group)
      launch_gemm(stream, o1B, wvB + (size_t)l*1048576, wv_b + (size_t)l*1024,
                  attF, nullptr, 1024, 64, 1024, 1024, 1024, 1024,
                  1048576, 65536, 64, 64, 16);

      // residual + relu
      if (l == 0) {
        combine_relu<<<4096, 256, 0, stream>>>(xiF, attF, xF0 + (size_t)i*1048576,
                                               xB0 + (size_t)i*1048576, 1048576);
      } else {
        combine_relu<<<4096, 256, 0, stream>>>(xiF, attF, out + (size_t)i*1048576,
                                               nullptr, 1048576);
      }
    }
    if (l == 0) {
      // fc2: x = x @ fc2_w + b over all 2048 rows
      launch_gemm(stream, xB0, w2T, fc2_b, xF1, xB1,
                  2048, 1024, 1024, 1024, 1024, 1024, 0, 0, 0, 0, 1);
    }
  }
}
